// EmotionGAT_21758304321918
// MI455X (gfx1250) — matrix operand order repaired and checked by reading, not compiled, not run on hardware
//
#include <hip/hip_runtime.h>
#include <hip/hip_bf16.h>
#include <math.h>
#include <type_traits>

typedef __attribute__((ext_vector_type(16))) _Float16 v16h;
typedef __attribute__((ext_vector_type(8)))  float    v8f;

#define GN 300000
#define GE 1500000
#define GB 50000

// ---------------- helpers ----------------
__device__ __forceinline__ unsigned fflip(float f) {
    unsigned u = __float_as_uint(f);
    return (u & 0x80000000u) ? ~u : (u | 0x80000000u);
}
__device__ __forceinline__ float funflip(unsigned u) {
    u = (u & 0x80000000u) ? (u & 0x7FFFFFFFu) : ~u;
    return __uint_as_float(u);
}
__device__ __forceinline__ float lrelu02(float x) { return x > 0.f ? x : 0.2f * x; }
__device__ __forceinline__ float eluf(float x)    { return x > 0.f ? x : expm1f(x); }

// ---------------- WMMA GEMM: out[M x 64] = X[M x KIN] @ W[KIN x 64] ----------------
// One wave -> 16 rows x 64 cols; 256 threads = 8 waves -> 128 rows / block.
// W staged to LDS transposed (column-major, K contiguous): B fragment = one 32B LDS
// read per lane. Compile-time KIN fully unrolls the K loop. FULL-tile fast path
// removes all bounds guards: stores use one base address + constant offsets.
template <int KIN>
__global__ __launch_bounds__(256)
void gemm_f16_wmma(const float* __restrict__ X, const float* __restrict__ W,
                   float* __restrict__ Out, int M) {
    constexpr int NKB  = (KIN + 31) / 32;
    constexpr int KPAD = NKB * 32;
    __shared__ _Float16 WlT[64 * KPAD];             // col-major: WlT[col*KPAD + k]
    const int tid = threadIdx.x;
    for (int i = tid; i < 64 * KPAD; i += 256) {
        int col = i / KPAD, k = i % KPAD;
        WlT[i] = (k < KIN) ? (_Float16)W[k * 64 + col] : (_Float16)0.f;
    }
    __syncthreads();

    const int wave = tid >> 5;
    const int lane = tid & 31;
    const int l16  = lane & 15;
    const bool hi  = lane >= 16;
    const int rowbase = (int)blockIdx.x * 128 + wave * 16;
    const int arow = rowbase + l16;
    const float* xr = X + (size_t)arow * KIN;

    auto body = [&](auto FULLC) {
        constexpr bool FULL = decltype(FULLC)::value;
        v8f acc[4];
        #pragma unroll
        for (int nt = 0; nt < 4; ++nt) acc[nt] = (v8f){};

        #pragma unroll
        for (int kb = 0; kb < NKB; ++kb) {
            // A fragment (16x32 f16): lanes<16 K{b+0..15}, lanes>=16 K{b+16..31}
            v16h a = (v16h){};
            const int kbase = kb * 32 + (hi ? 16 : 0);
            if (FULL || arow < M) {
                if (KIN % 32 == 0) {        // vectorized path (KIN = 64, 96)
                    const float* xp = xr + kbase;
                    float4 f0 = *(const float4*)(xp);
                    float4 f1 = *(const float4*)(xp + 4);
                    float4 f2 = *(const float4*)(xp + 8);
                    float4 f3 = *(const float4*)(xp + 12);
                    a[0]  = (_Float16)f0.x; a[1]  = (_Float16)f0.y; a[2]  = (_Float16)f0.z; a[3]  = (_Float16)f0.w;
                    a[4]  = (_Float16)f1.x; a[5]  = (_Float16)f1.y; a[6]  = (_Float16)f1.z; a[7]  = (_Float16)f1.w;
                    a[8]  = (_Float16)f2.x; a[9]  = (_Float16)f2.y; a[10] = (_Float16)f2.z; a[11] = (_Float16)f2.w;
                    a[12] = (_Float16)f3.x; a[13] = (_Float16)f3.y; a[14] = (_Float16)f3.z; a[15] = (_Float16)f3.w;
                } else {                    // ragged path (KIN = 12)
                    #pragma unroll
                    for (int t = 0; t < 8; ++t) {
                        int k0 = kbase + t;
                        int k1 = kbase + 8 + t;
                        a[t]     = (k0 < KIN) ? (_Float16)xr[k0] : (_Float16)0.f;
                        a[8 + t] = (k1 < KIN) ? (_Float16)xr[k1] : (_Float16)0.f;
                    }
                }
            }
            const int khalf = kb * 32 + (hi ? 16 : 0);
            #pragma unroll
            for (int nt = 0; nt < 4; ++nt) {
                const int col = nt * 16 + l16;
                v16h b = *(const v16h*)(&WlT[col * KPAD + khalf]);
                acc[nt] = __builtin_amdgcn_wmma_f32_16x16x32_f16(
                    false, a, false, b, (short)0, acc[nt], false, false);
            }
        }
        // C/D layout: VGPR j, lanes<16 -> row j, lanes>=16 -> row j+8; col = lane%16
        if (FULL) {
            // one base address per lane; all offsets constant -> folded into store offset
            float* po = Out + (size_t)rowbase * 64 + (hi ? 8 * 64 : 0) + l16;
            #pragma unroll
            for (int nt = 0; nt < 4; ++nt)
                #pragma unroll
                for (int j = 0; j < 8; ++j)
                    po[j * 64 + nt * 16] = acc[nt][j];
        } else {
            #pragma unroll
            for (int nt = 0; nt < 4; ++nt) {
                const int col = nt * 16 + l16;
                #pragma unroll
                for (int j = 0; j < 8; ++j) {
                    int row = rowbase + j + (hi ? 8 : 0);
                    if (row < M) Out[(size_t)row * 64 + col] = acc[nt][j];
                }
            }
        }
    };
    if (rowbase + 16 <= M) body(std::integral_constant<bool, true>{});
    else                   body(std::integral_constant<bool, false>{});
}

// ---------------- per-node attention logits ----------------
template <int H, int C>
__global__ void attn_logits(const float* __restrict__ Hbuf,
                            const float* __restrict__ a_s, const float* __restrict__ a_d,
                            float* __restrict__ als, float* __restrict__ ald, int Nn) {
    int i = blockIdx.x * blockDim.x + threadIdx.x;
    if (i >= Nn * H) return;
    int n = i / H, h = i % H;
    const float* hr = Hbuf + (size_t)n * H * C + h * C;
    float ss = 0.f, sd = 0.f;
    #pragma unroll
    for (int q = 0; q < C / 4; ++q) {
        float4 hv = *(const float4*)(hr + q * 4);
        float4 as4 = *(const float4*)(a_s + h * C + q * 4);
        float4 ad4 = *(const float4*)(a_d + h * C + q * 4);
        ss += hv.x * as4.x + hv.y * as4.y + hv.z * as4.z + hv.w * as4.w;
        sd += hv.x * ad4.x + hv.y * ad4.y + hv.z * ad4.z + hv.w * ad4.w;
    }
    als[i] = ss; ald[i] = sd;
}

// ---------------- edge-phase scratch init ----------------
__global__ void edge_init(unsigned* __restrict__ m, float* __restrict__ s,
                          float* __restrict__ agg, int nmh, int nagg) {
    int i = blockIdx.x * blockDim.x + threadIdx.x;
    if (i < nmh) { m[i] = 0x007FFFFFu; /* fflip(-inf) */ s[i] = 0.f; }
    if (i < nagg) agg[i] = 0.f;
}

// ---------------- pass 1: segment max of leaky-relu logits ----------------
template <int H>
__global__ void edge_pass1(const int* __restrict__ ei, int E, int Etot,
                           const float* __restrict__ als, const float* __restrict__ ald,
                           unsigned* __restrict__ m) {
    int i = blockIdx.x * blockDim.x + threadIdx.x;
    if (i >= Etot) return;
    int src = (i < E) ? ei[i]     : (i - E);
    int dst = (i < E) ? ei[E + i] : (i - E);
    #pragma unroll
    for (int h = 0; h < H; ++h) {
        float lg = lrelu02(als[src * H + h] + ald[dst * H + h]);
        atomicMax(&m[dst * H + h], fflip(lg));
    }
}

// ---------------- pass 2: segment sum of exp(logit - max) ----------------
template <int H>
__global__ void edge_pass2(const int* __restrict__ ei, int E, int Etot,
                           const float* __restrict__ als, const float* __restrict__ ald,
                           const unsigned* __restrict__ m, float* __restrict__ s) {
    int i = blockIdx.x * blockDim.x + threadIdx.x;
    if (i >= Etot) return;
    int src = (i < E) ? ei[i]     : (i - E);
    int dst = (i < E) ? ei[E + i] : (i - E);
    #pragma unroll
    for (int h = 0; h < H; ++h) {
        float lg = lrelu02(als[src * H + h] + ald[dst * H + h]);
        float ev = __expf(lg - funflip(m[dst * H + h]));
        atomicAdd(&s[dst * H + h], ev);
    }
}

// ---------------- pass 3: alpha-weighted scatter-add of source features ----------------
template <int H, int C>
__global__ void edge_pass3(const int* __restrict__ ei, int E, int Etot,
                           const float* __restrict__ als, const float* __restrict__ ald,
                           const unsigned* __restrict__ m, const float* __restrict__ s,
                           const float* __restrict__ Hbuf, float* __restrict__ agg) {
    int gid = blockIdx.x * blockDim.x + threadIdx.x;
    if (gid >= Etot * H) return;
    int e = gid / H, h = gid % H;
    int src = (e < E) ? ei[e]     : (e - E);
    int dst = (e < E) ? ei[E + e] : (e - E);
    float lg = lrelu02(als[src * H + h] + ald[dst * H + h]);
    float ev = __expf(lg - funflip(m[dst * H + h]));
    float alpha = ev / (s[dst * H + h] + 1e-16f);
    const float* hs = Hbuf + (size_t)src * H * C + h * C;
    float* ad = agg + (size_t)dst * H * C + h * C;
    #pragma unroll
    for (int q = 0; q < C / 4; ++q) {
        float4 hv = *(const float4*)(hs + q * 4);
        atomicAdd(&ad[q * 4 + 0], hv.x * alpha);
        atomicAdd(&ad[q * 4 + 1], hv.y * alpha);
        atomicAdd(&ad[q * 4 + 2], hv.z * alpha);
        atomicAdd(&ad[q * 4 + 3], hv.w * alpha);
    }
}

// ---------------- epilogue: bias + LayerNorm + ELU (+ optional residual) ----------------
template <int D, bool RESID>
__global__ void ep_ln_elu(const float* __restrict__ agg, const float* __restrict__ bias,
                          const float* __restrict__ g, const float* __restrict__ b,
                          const float* __restrict__ resid, float* __restrict__ out, int Nn) {
    int n = blockIdx.x * blockDim.x + threadIdx.x;
    if (n >= Nn) return;
    float v[D];
    float mu = 0.f;
    #pragma unroll
    for (int q = 0; q < D / 4; ++q) {
        float4 av = *(const float4*)(agg + (size_t)n * D + q * 4);
        float4 bv = *(const float4*)(bias + q * 4);
        v[q * 4 + 0] = av.x + bv.x; v[q * 4 + 1] = av.y + bv.y;
        v[q * 4 + 2] = av.z + bv.z; v[q * 4 + 3] = av.w + bv.w;
        mu += v[q * 4 + 0] + v[q * 4 + 1] + v[q * 4 + 2] + v[q * 4 + 3];
    }
    mu *= (1.f / (float)D);
    float var = 0.f;
    #pragma unroll
    for (int d = 0; d < D; ++d) { float t = v[d] - mu; var += t * t; }
    var *= (1.f / (float)D);
    float rs = rsqrtf(var + 1e-5f);
    #pragma unroll
    for (int q = 0; q < D / 4; ++q) {
        float4 gv = *(const float4*)(g + q * 4);
        float4 bb = *(const float4*)(b + q * 4);
        float4 y;
        y.x = eluf((v[q * 4 + 0] - mu) * rs * gv.x + bb.x);
        y.y = eluf((v[q * 4 + 1] - mu) * rs * gv.y + bb.y);
        y.z = eluf((v[q * 4 + 2] - mu) * rs * gv.z + bb.z);
        y.w = eluf((v[q * 4 + 3] - mu) * rs * gv.w + bb.w);
        if (RESID) {
            float4 rv = *(const float4*)(resid + (size_t)n * D + q * 4);
            y.x += rv.x; y.y += rv.y; y.z += rv.z; y.w += rv.w;
        }
        *(float4*)(out + (size_t)n * D + q * 4) = y;
    }
}

// layer-3 epilogue: mean over 2 heads (64 -> 32), bias, LN(32), ELU
__global__ void ep_headmean_ln_elu(const float* __restrict__ agg, const float* __restrict__ bias,
                                   const float* __restrict__ g, const float* __restrict__ b,
                                   float* __restrict__ out, int Nn) {
    int n = blockIdx.x * blockDim.x + threadIdx.x;
    if (n >= Nn) return;
    float v[32];
    float mu = 0.f;
    #pragma unroll
    for (int q = 0; q < 8; ++q) {
        float4 h0 = *(const float4*)(agg + (size_t)n * 64 + q * 4);
        float4 h1 = *(const float4*)(agg + (size_t)n * 64 + 32 + q * 4);
        float4 bv = *(const float4*)(bias + q * 4);
        v[q * 4 + 0] = 0.5f * (h0.x + h1.x) + bv.x;
        v[q * 4 + 1] = 0.5f * (h0.y + h1.y) + bv.y;
        v[q * 4 + 2] = 0.5f * (h0.z + h1.z) + bv.z;
        v[q * 4 + 3] = 0.5f * (h0.w + h1.w) + bv.w;
        mu += v[q * 4 + 0] + v[q * 4 + 1] + v[q * 4 + 2] + v[q * 4 + 3];
    }
    mu *= (1.f / 32.f);
    float var = 0.f;
    #pragma unroll
    for (int d = 0; d < 32; ++d) { float t = v[d] - mu; var += t * t; }
    var *= (1.f / 32.f);
    float rs = rsqrtf(var + 1e-5f);
    #pragma unroll
    for (int q = 0; q < 8; ++q) {
        float4 gv = *(const float4*)(g + q * 4);
        float4 bb = *(const float4*)(b + q * 4);
        float4 y;
        y.x = eluf((v[q * 4 + 0] - mu) * rs * gv.x + bb.x);
        y.y = eluf((v[q * 4 + 1] - mu) * rs * gv.y + bb.y);
        y.z = eluf((v[q * 4 + 2] - mu) * rs * gv.z + bb.z);
        y.w = eluf((v[q * 4 + 3] - mu) * rs * gv.w + bb.w);
        *(float4*)(out + (size_t)n * 32 + q * 4) = y;
    }
}

// aggregator epilogue: bias + LN(64) + ReLU (in-place safe: thread owns row)
__global__ void ep_ln_relu(float* __restrict__ a, const float* __restrict__ bias,
                           const float* __restrict__ g, const float* __restrict__ b, int Nn) {
    int n = blockIdx.x * blockDim.x + threadIdx.x;
    if (n >= Nn) return;
    float v[64];
    float mu = 0.f;
    #pragma unroll
    for (int q = 0; q < 16; ++q) {
        float4 av = *(const float4*)(a + (size_t)n * 64 + q * 4);
        float4 bv = *(const float4*)(bias + q * 4);
        v[q * 4 + 0] = av.x + bv.x; v[q * 4 + 1] = av.y + bv.y;
        v[q * 4 + 2] = av.z + bv.z; v[q * 4 + 3] = av.w + bv.w;
        mu += v[q * 4 + 0] + v[q * 4 + 1] + v[q * 4 + 2] + v[q * 4 + 3];
    }
    mu *= (1.f / 64.f);
    float var = 0.f;
    #pragma unroll
    for (int d = 0; d < 64; ++d) { float t = v[d] - mu; var += t * t; }
    var *= (1.f / 64.f);
    float rs = rsqrtf(var + 1e-5f);
    #pragma unroll
    for (int q = 0; q < 16; ++q) {
        float4 gv = *(const float4*)(g + q * 4);
        float4 bb = *(const float4*)(b + q * 4);
        float4 y;
        y.x = fmaxf((v[q * 4 + 0] - mu) * rs * gv.x + bb.x, 0.f);
        y.y = fmaxf((v[q * 4 + 1] - mu) * rs * gv.y + bb.y, 0.f);
        y.z = fmaxf((v[q * 4 + 2] - mu) * rs * gv.z + bb.z, 0.f);
        y.w = fmaxf((v[q * 4 + 3] - mu) * rs * gv.w + bb.w, 0.f);
        *(float4*)(a + (size_t)n * 64 + q * 4) = y;
    }
}

// ---------------- per-graph emotion MLPs: (6 x 32) -> stack[96] ----------------
__global__ void emomlp(const float* __restrict__ x3,
                       const float* __restrict__ W1, const float* __restrict__ b1,
                       const float* __restrict__ lg, const float* __restrict__ lb,
                       const float* __restrict__ W2, const float* __restrict__ b2,
                       float* __restrict__ stack, int B) {
    int gidx = blockIdx.x * blockDim.x + threadIdx.x;
    if (gidx >= B) return;
    const float* ef = x3 + (size_t)gidx * 6 * 32;
    for (int i = 0; i < 6; ++i) {
        float h[16];
        for (int k = 0; k < 16; ++k) {
            float s = b1[i * 16 + k];
            for (int d = 0; d < 32; ++d)
                s += ef[i * 32 + d] * W1[(i * 32 + d) * 16 + k];
            h[k] = s;
        }
        float mu = 0.f;
        for (int k = 0; k < 16; ++k) mu += h[k];
        mu *= (1.f / 16.f);
        float var = 0.f;
        for (int k = 0; k < 16; ++k) { float t = h[k] - mu; var += t * t; }
        var *= (1.f / 16.f);
        float rs = rsqrtf(var + 1e-5f);
        float t2[16];
        for (int k = 0; k < 16; ++k) {
            float y = (h[k] - mu) * rs * lg[i * 16 + k] + lb[i * 16 + k];
            t2[k] = y > 0.f ? y : 0.f;
        }
        for (int k2 = 0; k2 < 16; ++k2) {
            float s = b2[i * 16 + k2];
            for (int k = 0; k < 16; ++k)
                s += t2[k] * W2[(i * 16 + k) * 16 + k2];
            stack[(size_t)gidx * 96 + i * 16 + k2] = s;
        }
    }
}

// ---------------- final head: raw = a@W2+b2; inter-softmax mix; temperature ----------------
__global__ void finalize(const float* __restrict__ a, const float* __restrict__ W2,
                         const float* __restrict__ b2, const float* __restrict__ inter,
                         const float* __restrict__ temp, float* __restrict__ out, int B) {
    int gidx = blockIdx.x * blockDim.x + threadIdx.x;
    if (gidx >= B) return;
    float raw[6];
    for (int j = 0; j < 6; ++j) {
        float s = b2[j];
        for (int k = 0; k < 64; ++k) s += a[(size_t)gidx * 64 + k] * W2[k * 6 + j];
        raw[j] = s;
    }
    float iw[36];
    for (int r = 0; r < 6; ++r) {
        float mx = inter[r * 6];
        for (int c = 1; c < 6; ++c) mx = fmaxf(mx, inter[r * 6 + c]);
        float ssum = 0.f;
        for (int c = 0; c < 6; ++c) { float e = __expf(inter[r * 6 + c] - mx); iw[r * 6 + c] = e; ssum += e; }
        float inv = 1.f / ssum;
        for (int c = 0; c < 6; ++c) iw[r * 6 + c] *= inv;
    }
    float invt = 1.f / fmaxf(temp[0], 0.1f);
    for (int j = 0; j < 6; ++j) {
        float interacted = 0.f;
        for (int r = 0; r < 6; ++r) interacted += raw[r] * iw[r * 6 + j];
        out[(size_t)gidx * 6 + j] = (raw[j] + interacted) * 0.5f * invt;
    }
}

// ---------------- host-side per-layer orchestration ----------------
template <int KIN, int H, int C>
static inline void launch_gat_layer(const float* xin,
                                    const float* W, const float* as, const float* ad,
                                    const int* ei,
                                    float* hb, float* als, float* ald,
                                    unsigned* mfl, float* ssum, float* agg,
                                    hipStream_t stream) {
    const int N = GN, E = GE, Etot = GE + GN;
    gemm_f16_wmma<KIN><<<dim3((N + 127) / 128), dim3(256), 0, stream>>>(xin, W, hb, N);
    attn_logits<H, C><<<dim3((N * H + 255) / 256), dim3(256), 0, stream>>>(hb, as, ad, als, ald, N);
    edge_init<<<dim3((N * 64 + 255) / 256), dim3(256), 0, stream>>>(mfl, ssum, agg, N * H, N * 64);
    edge_pass1<H><<<dim3((Etot + 255) / 256), dim3(256), 0, stream>>>(ei, E, Etot, als, ald, mfl);
    edge_pass2<H><<<dim3((Etot + 255) / 256), dim3(256), 0, stream>>>(ei, E, Etot, als, ald, mfl, ssum);
    edge_pass3<H, C><<<dim3((Etot * H + 255) / 256), dim3(256), 0, stream>>>(ei, E, Etot, als, ald, mfl, ssum, hb, agg);
}

extern "C" void kernel_launch(void* const* d_in, const int* in_sizes, int n_in,
                              void* d_out, int out_size, void* d_ws, size_t ws_size,
                              hipStream_t stream) {
    (void)in_sizes; (void)n_in; (void)out_size; (void)ws_size;
    const float* x     = (const float*)d_in[0];
    const int*   ei    = (const int*)d_in[1];
    const float* g1_W  = (const float*)d_in[3];
    const float* g1_as = (const float*)d_in[4];
    const float* g1_ad = (const float*)d_in[5];
    const float* g1_b  = (const float*)d_in[6];
    const float* g2_W  = (const float*)d_in[7];
    const float* g2_as = (const float*)d_in[8];
    const float* g2_ad = (const float*)d_in[9];
    const float* g2_b  = (const float*)d_in[10];
    const float* g3_W  = (const float*)d_in[11];
    const float* g3_as = (const float*)d_in[12];
    const float* g3_ad = (const float*)d_in[13];
    const float* g3_b  = (const float*)d_in[14];
    const float* ln1_g = (const float*)d_in[15];
    const float* ln1_b = (const float*)d_in[16];
    const float* ln2_g = (const float*)d_in[17];
    const float* ln2_b = (const float*)d_in[18];
    const float* ln3_g = (const float*)d_in[19];
    const float* ln3_b = (const float*)d_in[20];
    const float* etW1  = (const float*)d_in[21];
    const float* etb1  = (const float*)d_in[22];
    const float* etlg  = (const float*)d_in[23];
    const float* etlb  = (const float*)d_in[24];
    const float* etW2  = (const float*)d_in[25];
    const float* etb2  = (const float*)d_in[26];
    const float* agW1  = (const float*)d_in[27];
    const float* agb1  = (const float*)d_in[28];
    const float* aglg  = (const float*)d_in[29];
    const float* aglb  = (const float*)d_in[30];
    const float* agW2  = (const float*)d_in[31];
    const float* agb2  = (const float*)d_in[32];
    const float* inter = (const float*)d_in[33];
    const float* temp  = (const float*)d_in[34];
    float* out = (float*)d_out;

    // workspace carve-up (~326 MB)
    float* ws = (float*)d_ws;
    size_t off = 0;
    float* x1  = ws + off; off += (size_t)GN * 64;
    float* x2  = ws + off; off += (size_t)GN * 64;
    float* hb  = ws + off; off += (size_t)GN * 64;   // reused as stack[B,96] later
    float* agg = ws + off; off += (size_t)GN * 64;   // reused as a[B,64] later
    float* als = ws + off; off += (size_t)GN * 4;
    float* ald = ws + off; off += (size_t)GN * 4;
    float* ssum = ws + off; off += (size_t)GN * 4;
    unsigned* mfl = (unsigned*)(ws + off); off += (size_t)GN * 4;

    const int N = GN, B = GB;

    // ---- GAT layer 1: 12 -> 4x16 concat ----
    launch_gat_layer<12, 4, 16>(x, g1_W, g1_as, g1_ad, ei, hb, als, ald, mfl, ssum, agg, stream);
    ep_ln_elu<64, false><<<dim3((N + 255) / 256), dim3(256), 0, stream>>>(agg, g1_b, ln1_g, ln1_b, nullptr, x1, N);

    // ---- GAT layer 2: 64 -> 4x16 concat, residual x1 ----
    launch_gat_layer<64, 4, 16>(x1, g2_W, g2_as, g2_ad, ei, hb, als, ald, mfl, ssum, agg, stream);
    ep_ln_elu<64, true><<<dim3((N + 255) / 256), dim3(256), 0, stream>>>(agg, g2_b, ln2_g, ln2_b, x1, x2, N);

    // ---- GAT layer 3: 64 -> 2x32 mean ----
    launch_gat_layer<64, 2, 32>(x2, g3_W, g3_as, g3_ad, ei, hb, als, ald, mfl, ssum, agg, stream);
    float* x3 = x1; // N x 32, reuse
    ep_headmean_ln_elu<<<dim3((N + 255) / 256), dim3(256), 0, stream>>>(agg, g3_b, ln3_g, ln3_b, x3, N);

    // ---- per-graph emotion MLPs -> stack[B,96] ----
    float* stack = hb;
    emomlp<<<dim3((B + 255) / 256), dim3(256), 0, stream>>>(x3, etW1, etb1, etlg, etlb, etW2, etb2, stack, B);

    // ---- aggregator GEMM (WMMA): [B,96] @ [96,64] -> a[B,64]; LN+ReLU ----
    float* abuf = agg;
    gemm_f16_wmma<96><<<dim3((B + 127) / 128), dim3(256), 0, stream>>>(stack, agW1, abuf, B);
    ep_ln_relu<<<dim3((B + 255) / 256), dim3(256), 0, stream>>>(abuf, agb1, aglg, aglb, B);

    // ---- final head ----
    finalize<<<dim3((B + 255) / 256), dim3(256), 0, stream>>>(abuf, agW2, agb2, inter, temp, out, B);
}
